// TransformerBlock_4801773437198
// MI455X (gfx1250) — matrix operand order repaired and checked by reading, not compiled, not run on hardware
//
#include <hip/hip_runtime.h>
#include <hip/hip_bf16.h>
#include <math.h>

typedef __bf16 bf16;
typedef __attribute__((ext_vector_type(16))) __bf16 bf16x16;
typedef __attribute__((ext_vector_type(8)))  float  f32x8;

#define D_MODEL 1024
#define NUM_HEADS 16
#define D_HEAD 64
#define D_MLP 4096
#define SEQ 2048
#define BATCH 4
#define BS (BATCH * SEQ) // 8192

// ---------------------------------------------------------------------------
// Async global->LDS (CDNA5 GLOBAL_LOAD_ASYNC_TO_LDS_B128, ASYNCcnt-tracked).
// Signature: (v4i __device__*, v4i __shared__*, imm int offset, imm int cpol)
// Guarded: falls back to register-staged copies if the builtin is absent.
// ---------------------------------------------------------------------------
#if defined(__HIP_DEVICE_COMPILE__) && __has_builtin(__builtin_amdgcn_global_load_async_to_lds_b128)
#define ASYNC_LDS 1
#else
#define ASYNC_LDS 0
#endif

#if ASYNC_LDS
typedef int v4i_b128 __attribute__((vector_size(4 * sizeof(int))));
typedef __attribute__((address_space(1))) v4i_b128* gptr_b128;
typedef __attribute__((address_space(3))) v4i_b128* lptr_b128;

template <int OFF>
__device__ __forceinline__ void async_ld16(bf16* lds, const bf16* g) {
  __builtin_amdgcn_global_load_async_to_lds_b128((gptr_b128)g, (lptr_b128)lds, OFF, 0);
}
__device__ __forceinline__ void async_ld64B(bf16* lds, const bf16* g) {
  async_ld16<0>(lds, g);
  async_ld16<16>(lds, g);
  async_ld16<32>(lds, g);
  async_ld16<48>(lds, g);
}
__device__ __forceinline__ void async_wait0() {
#if __has_builtin(__builtin_amdgcn_s_wait_asynccnt)
  __builtin_amdgcn_s_wait_asynccnt(0);
#else
  asm volatile("s_wait_asynccnt 0" ::: "memory");
#endif
}
#endif

// ---------------------------------------------------------------------------
// WMMA fragment helpers (wave32; A and B both use contiguous k = 16*(lane/16) + s)
// ---------------------------------------------------------------------------
__device__ __forceinline__ f32x8 wmma_bf16(const bf16x16& a, const bf16x16& b, const f32x8& c) {
  return __builtin_amdgcn_wmma_f32_16x16x32_bf16(false, a, false, b, (short)0, c, false, false);
}

// A-matrix 16x32 bf16 from LDS tile stored [row][k]; 2x ds_load_b128 per lane.
__device__ __forceinline__ void frag_load_a(const bf16* s, int row0, int k0, int ld, bf16x16& a) {
  int lane = threadIdx.x & 31;
  int l = lane & 15, hi = lane >> 4;
  const bf16* base = s + (row0 + l) * ld + k0;
#pragma unroll
  for (int g = 0; g < 8; ++g) {
    int kb = 2 * g + hi * 16;
    a[2 * g]     = base[kb];
    a[2 * g + 1] = base[kb + 1];
  }
}

// B-matrix 32x16 (KxN) where LDS tile is stored [n][k]: contiguous 16 elements.
__device__ __forceinline__ void frag_load_b_nk(const bf16* s, int n0, int k0, int ld, bf16x16& b) {
  int lane = threadIdx.x & 31;
  int l = lane & 15, hi = lane >> 4;
  const bf16* base = s + (n0 + l) * ld + k0 + hi * 16;
#pragma unroll
  for (int e = 0; e < 16; ++e) b[e] = base[e];
}

// ---------------------------------------------------------------------------
// f32 -> bf16 conversion, 8 elements / thread
// ---------------------------------------------------------------------------
__global__ void cvt_f32_bf16(const float* __restrict__ in, bf16* __restrict__ out, int n8) {
  int i = blockIdx.x * blockDim.x + threadIdx.x;
  if (i < n8) {
    const float4* s = (const float4*)in + (size_t)i * 2;
    float4 f0 = s[0], f1 = s[1];
    bf16 v[8] = {(bf16)f0.x, (bf16)f0.y, (bf16)f0.z, (bf16)f0.w,
                 (bf16)f1.x, (bf16)f1.y, (bf16)f1.z, (bf16)f1.w};
    *(uint4*)(out + (size_t)i * 8) = *(const uint4*)v;
  }
}

// ---------------------------------------------------------------------------
// bf16 GEMM:  C[M][ldc] = A[M][K] * B[N][K]^T  (+bias, gelu, residual)
// 128x128 tile, BK=64, ping-pong double-buffered LDS, async fills.
// ---------------------------------------------------------------------------
#define BM 128
#define BN 128
#define BKT 64
#define GLD 72 // 64 + 8 pad; 144-byte rows keep 16B alignment

template <bool HAS_BIAS, bool GELU_ACT, bool HAS_RES, bool OUT_F32, bool OUT_BF16>
__global__ __launch_bounds__(256) void gemm_bf16_nt(
    const bf16* __restrict__ A, int K,
    const bf16* __restrict__ B,
    const float* __restrict__ bias,   // [N]
    const float* __restrict__ resid,  // [M][ldc] f32
    float* __restrict__ Cf,
    bf16* __restrict__ Cb,
    int ldc) {
  __shared__ bf16 As[2][BM * GLD];
  __shared__ bf16 Bs[2][BN * GLD];

  int tid = threadIdx.x;
  int wid = tid >> 5;
  int lane = tid & 31, l = lane & 15, hi = lane >> 4;
  int bm = blockIdx.y * BM;
  int bn = blockIdx.x * BN;
  int wm = (wid & 3) * 32;
  int wn = (wid >> 2) * 64;

  f32x8 acc[2][4];
#pragma unroll
  for (int i = 0; i < 2; ++i)
#pragma unroll
    for (int n = 0; n < 4; ++n)
#pragma unroll
      for (int e = 0; e < 8; ++e) acc[i][n][e] = 0.f;

  int trow = tid >> 1;
  int tcol = (tid & 1) * 32; // 32 bf16 = 64B per thread per tile

  const bf16* arow_p = A + (size_t)(bm + trow) * K + tcol;
  const bf16* brow_p = B + (size_t)(bn + trow) * K + tcol;
  bf16* la0 = &As[0][trow * GLD + tcol];
  bf16* la1 = &As[1][trow * GLD + tcol];
  bf16* lb0 = &Bs[0][trow * GLD + tcol];
  bf16* lb1 = &Bs[1][trow * GLD + tcol];

#if ASYNC_LDS
  async_ld64B(la0, arow_p);
  async_ld64B(lb0, brow_p);
  async_wait0();
#else
  {
    const uint4* sa = (const uint4*)arow_p;
    uint4* da = (uint4*)la0;
    da[0] = sa[0]; da[1] = sa[1]; da[2] = sa[2]; da[3] = sa[3];
    const uint4* sb = (const uint4*)brow_p;
    uint4* db = (uint4*)lb0;
    db[0] = sb[0]; db[1] = sb[1]; db[2] = sb[2]; db[3] = sb[3];
  }
#endif
  __syncthreads();

  int nk = K / BKT;
  for (int kt = 0; kt < nk; ++kt) {
    int cur = kt & 1;
    bool more = (kt + 1 < nk);
#if ASYNC_LDS
    if (more) { // async fill of the free buffer; overlaps WMMA below
      async_ld64B(cur ? la0 : la1, arow_p + (kt + 1) * BKT);
      async_ld64B(cur ? lb0 : lb1, brow_p + (kt + 1) * BKT);
      if (kt + 2 < nk) {
        __builtin_prefetch(arow_p + (kt + 2) * BKT, 0, 0);
        __builtin_prefetch(brow_p + (kt + 2) * BKT, 0, 0);
      }
    }
#else
    uint4 pa[4], pb[4];
    if (more) {
      const uint4* sa = (const uint4*)(arow_p + (kt + 1) * BKT);
      pa[0] = sa[0]; pa[1] = sa[1]; pa[2] = sa[2]; pa[3] = sa[3];
      const uint4* sb = (const uint4*)(brow_p + (kt + 1) * BKT);
      pb[0] = sb[0]; pb[1] = sb[1]; pb[2] = sb[2]; pb[3] = sb[3];
      if (kt + 2 < nk) {
        __builtin_prefetch(arow_p + (kt + 2) * BKT, 0, 0);
        __builtin_prefetch(brow_p + (kt + 2) * BKT, 0, 0);
      }
    }
#endif
#pragma unroll
    for (int kc = 0; kc < BKT; kc += 32) {
      bf16x16 a0, a1;
      frag_load_a(&As[cur][0], wm, kc, GLD, a0);
      frag_load_a(&As[cur][0], wm + 16, kc, GLD, a1);
#pragma unroll
      for (int n = 0; n < 4; ++n) {
        bf16x16 b;
        frag_load_b_nk(&Bs[cur][0], wn + n * 16, kc, GLD, b);
        acc[0][n] = wmma_bf16(a0, b, acc[0][n]);
        acc[1][n] = wmma_bf16(a1, b, acc[1][n]);
      }
    }
#if ASYNC_LDS
    if (more) async_wait0();
#else
    if (more) {
      uint4* da = (uint4*)(cur ? la0 : la1);
      da[0] = pa[0]; da[1] = pa[1]; da[2] = pa[2]; da[3] = pa[3];
      uint4* db = (uint4*)(cur ? lb0 : lb1);
      db[0] = pb[0]; db[1] = pb[1]; db[2] = pb[2]; db[3] = pb[3];
    }
#endif
    __syncthreads();
  }

  float bias_v[4];
  if constexpr (HAS_BIAS) {
#pragma unroll
    for (int n = 0; n < 4; ++n) bias_v[n] = bias[bn + wn + n * 16 + l];
  }

#pragma unroll
  for (int i = 0; i < 2; ++i)
#pragma unroll
    for (int n = 0; n < 4; ++n)
#pragma unroll
      for (int r = 0; r < 8; ++r) {
        int row = bm + wm + i * 16 + r + hi * 8;
        int col = bn + wn + n * 16 + l;
        float v = acc[i][n][r];
        if constexpr (HAS_BIAS) v += bias_v[n];
        if constexpr (GELU_ACT) v = 0.5f * v * (1.0f + erff(v * 0.70710678118f));
        size_t idx = (size_t)row * ldc + col;
        if constexpr (HAS_RES) v += resid[idx];
        if constexpr (OUT_F32) Cf[idx] = v;
        if constexpr (OUT_BF16) Cb[idx] = (bf16)v;
      }
}

// ---------------------------------------------------------------------------
// Flash attention: one block = (batch, head, 64-query tile), 128 threads.
// qkv layout: [BS][3072], columns = [K(0..1023) | Q(1024..2047) | V(2048..)].
// K tiles [key][dh] (async-filled); V tiles TRANSPOSED [dh][key] via regs.
// ---------------------------------------------------------------------------
__global__ __launch_bounds__(128) void flash_attn(
    const bf16* __restrict__ qkv, bf16* __restrict__ z) {
  __shared__ bf16 Qt[64 * 72];
  __shared__ bf16 Kt[2][64 * 72];
  __shared__ bf16 VtT[2][64 * 72]; // [dh][key]
  __shared__ float Sf[64 * 68];
  __shared__ bf16 Pt[64 * 72];
  __shared__ float m_s[64], l_s[64], alpha_s[64];

  int tid = threadIdx.x;
  int wid = tid >> 5;
  int lane = tid & 31, l = lane & 15, hi = lane >> 4;
  int qt = blockIdx.x;
  int bh = blockIdx.y;
  int b = bh >> 4, h = bh & 15;
  int qbase = qt * 64;
  size_t rowbase = (size_t)b * SEQ * 3072;

  int r2 = tid >> 1, c2 = (tid & 1) * 32; // tile-load coords: key row, dh range

  { // load Q tile + K/V tile 0
    const bf16* gq = qkv + rowbase + (size_t)(qbase + r2) * 3072 + 1024 + h * 64 + c2;
    const bf16* gk = qkv + rowbase + (size_t)r2 * 3072 + h * 64 + c2;
    bf16* lq = Qt + r2 * 72 + c2;
    bf16* lk = &Kt[0][r2 * 72 + c2];
#if ASYNC_LDS
    async_ld64B(lq, gq);
    async_ld64B(lk, gk);
#else
    const uint4* sq = (const uint4*)gq;
    uint4* dq = (uint4*)lq;
    dq[0] = sq[0]; dq[1] = sq[1]; dq[2] = sq[2]; dq[3] = sq[3];
    const uint4* sk = (const uint4*)gk;
    uint4* dk = (uint4*)lk;
    dk[0] = sk[0]; dk[1] = sk[1]; dk[2] = sk[2]; dk[3] = sk[3];
#endif
    const uint4* sv = (const uint4*)(qkv + rowbase + (size_t)r2 * 3072 + 2048 + h * 64 + c2);
    uint4 vr[4];
    vr[0] = sv[0]; vr[1] = sv[1]; vr[2] = sv[2]; vr[3] = sv[3];
    const bf16* vh = (const bf16*)vr;
#pragma unroll
    for (int j = 0; j < 32; ++j) VtT[0][(c2 + j) * 72 + r2] = vh[j];
#if ASYNC_LDS
    async_wait0();
#endif
  }
  if (tid < 64) { m_s[tid] = -1e30f; l_s[tid] = 0.f; }
  __syncthreads();

  bf16x16 qa0, qa1; // this wave's 16 query rows, dh chunks 0..31 / 32..63
  frag_load_a(Qt, wid * 16, 0, 72, qa0);
  frag_load_a(Qt, wid * 16, 32, 72, qa1);

  f32x8 o[4];
#pragma unroll
  for (int n = 0; n < 4; ++n)
#pragma unroll
    for (int e = 0; e < 8; ++e) o[n][e] = 0.f;

  for (int kt = 0; kt <= qt; ++kt) {
    int kbase = kt * 64;
    int cur = kt & 1, nxt = cur ^ 1;
    bool more = (kt < qt);

    uint4 va[4];
    if (more) { // next K tile: async straight to LDS; V staged for transpose
#if ASYNC_LDS
      async_ld64B(&Kt[nxt][r2 * 72 + c2],
                  qkv + rowbase + (size_t)(kbase + 64 + r2) * 3072 + h * 64 + c2);
#else
      const uint4* sk = (const uint4*)(qkv + rowbase + (size_t)(kbase + 64 + r2) * 3072 + h * 64 + c2);
      uint4 ka[4];
      ka[0] = sk[0]; ka[1] = sk[1]; ka[2] = sk[2]; ka[3] = sk[3];
      uint4* dk = (uint4*)(&Kt[nxt][r2 * 72 + c2]);
      dk[0] = ka[0]; dk[1] = ka[1]; dk[2] = ka[2]; dk[3] = ka[3];
#endif
      const uint4* sv = (const uint4*)(qkv + rowbase + (size_t)(kbase + 64 + r2) * 3072 + 2048 + h * 64 + c2);
      va[0] = sv[0]; va[1] = sv[1]; va[2] = sv[2]; va[3] = sv[3];
    }

    // S = Q @ K^T (scaled, causal-masked) -> LDS
#pragma unroll
    for (int n = 0; n < 4; ++n) {
      f32x8 s;
#pragma unroll
      for (int e = 0; e < 8; ++e) s[e] = 0.f;
      bf16x16 kb0, kb1;
      frag_load_b_nk(&Kt[cur][0], n * 16, 0, 72, kb0);
      frag_load_b_nk(&Kt[cur][0], n * 16, 32, 72, kb1);
      s = wmma_bf16(qa0, kb0, s);
      s = wmma_bf16(qa1, kb1, s);
#pragma unroll
      for (int r = 0; r < 8; ++r) {
        int qrow = wid * 16 + r + hi * 8;
        int qg = qbase + qrow;
        int kg = kbase + n * 16 + l;
        Sf[qrow * 68 + n * 16 + l] = (kg <= qg) ? s[r] * 0.125f : -1e30f;
      }
    }
    __syncthreads();

    // online softmax: 2 threads per row, pair-combined with wave shuffles
    {
      int srow = tid >> 1, shalf = tid & 1;
      float mold = m_s[srow];
      const float* sr = &Sf[srow * 68 + shalf * 32];
      float pm = -1e30f;
#pragma unroll 8
      for (int c = 0; c < 32; ++c) pm = fmaxf(pm, sr[c]);
      pm = fmaxf(pm, __shfl_xor(pm, 1, 32));
      float mx = fmaxf(mold, pm);
      float sum = 0.f;
      bf16* pr = &Pt[srow * 72 + shalf * 32];
#pragma unroll 8
      for (int c = 0; c < 32; ++c) {
        float p = __expf(sr[c] - mx);
        sum += p;
        pr[c] = (bf16)p;
      }
      sum += __shfl_xor(sum, 1, 32);
      if (shalf == 0) {
        float alpha = __expf(mold - mx);
        m_s[srow] = mx;
        l_s[srow] = l_s[srow] * alpha + sum;
        alpha_s[srow] = alpha;
      }
    }
    __syncthreads();

    // O = O*alpha + P @ V ; commit transposed V to the other buffer
    float al[8];
#pragma unroll
    for (int r = 0; r < 8; ++r) al[r] = alpha_s[wid * 16 + r + hi * 8];

    bf16x16 pa0, pa1;
    frag_load_a(Pt, wid * 16, 0, 72, pa0);
    frag_load_a(Pt, wid * 16, 32, 72, pa1);
#pragma unroll
    for (int n = 0; n < 4; ++n) {
#pragma unroll
      for (int r = 0; r < 8; ++r) o[n][r] *= al[r];
      bf16x16 vb0, vb1;
      frag_load_b_nk(&VtT[cur][0], n * 16, 0, 72, vb0);  // N=dh tile, keys 0..31
      frag_load_b_nk(&VtT[cur][0], n * 16, 32, 72, vb1); // keys 32..63
      o[n] = wmma_bf16(pa0, vb0, o[n]);
      o[n] = wmma_bf16(pa1, vb1, o[n]);
    }

    if (more) {
      const bf16* vh = (const bf16*)va;
#pragma unroll
      for (int j = 0; j < 32; ++j) VtT[nxt][(c2 + j) * 72 + r2] = vh[j];
#if ASYNC_LDS
      async_wait0();
#endif
    }
    __syncthreads();
  }

  // normalize and write z
  float linv[8];
#pragma unroll
  for (int r = 0; r < 8; ++r) linv[r] = 1.0f / l_s[wid * 16 + r + hi * 8];
#pragma unroll
  for (int n = 0; n < 4; ++n)
#pragma unroll
    for (int r = 0; r < 8; ++r) {
      int qrow = wid * 16 + r + hi * 8;
      float v = o[n][r] * linv[r];
      z[(size_t)(b * SEQ + qbase + qrow) * 1024 + h * 64 + n * 16 + l] = (bf16)v;
    }
}

// ---------------------------------------------------------------------------
// Launcher
// ---------------------------------------------------------------------------
extern "C" void kernel_launch(void* const* d_in, const int* in_sizes, int n_in,
                              void* d_out, int out_size, void* d_ws, size_t ws_size,
                              hipStream_t stream) {
  const float* x     = (const float*)d_in[0];
  const float* W_K   = (const float*)d_in[1];
  const float* W_Q   = (const float*)d_in[2];
  const float* W_V   = (const float*)d_in[3];
  const float* W_O   = (const float*)d_in[4];
  const float* W_in  = (const float*)d_in[5];
  const float* b_in  = (const float*)d_in[6];
  const float* W_out = (const float*)d_in[7];
  const float* b_out = (const float*)d_in[8];
  float* out = (float*)d_out;

  char* ws = (char*)d_ws;
  size_t off = 0;
  auto alloc = [&](size_t bytes) -> char* {
    char* p = ws + off;
    off += (bytes + 255) & ~(size_t)255;
    return p;
  };

  bf16* wqkv = (bf16*)alloc((size_t)3072 * 1024 * 2); // rows: [K | Q | V]
  bf16* wo   = (bf16*)alloc((size_t)1024 * 1024 * 2);
  bf16* win  = (bf16*)alloc((size_t)4096 * 1024 * 2);
  bf16* wout = (bf16*)alloc((size_t)1024 * 4096 * 2);
  bf16* xb   = (bf16*)alloc((size_t)BS * 1024 * 2);   // x bf16; reused as z
  char* big  = alloc((size_t)BS * D_MLP * 2);         // qkv (50MB) then h (67MB)
  float* x1f = (float*)alloc((size_t)BS * 1024 * 4);  // x + attn_out, f32
  bf16* x1b  = (bf16*)alloc((size_t)BS * 1024 * 2);   // same in bf16

  bf16* qkv = (bf16*)big;
  bf16* hb  = (bf16*)big; // alias: qkv dead before MLP hidden is written
  bf16* zb  = xb;         // alias: xb dead after the QKV GEMM

  auto cvt = [&](const float* src, bf16* dst, size_t n) {
    size_t n8 = n / 8;
    cvt_f32_bf16<<<(unsigned)((n8 + 255) / 256), 256, 0, stream>>>(src, dst, (int)n8);
  };
  cvt(W_K, wqkv, (size_t)1024 * 1024);
  cvt(W_Q, wqkv + (size_t)1024 * 1024, (size_t)1024 * 1024);
  cvt(W_V, wqkv + (size_t)2048 * 1024, (size_t)1024 * 1024);
  cvt(W_O, wo, (size_t)1024 * 1024);
  cvt(W_in, win, (size_t)4096 * 1024);
  cvt(W_out, wout, (size_t)4096 * 1024);
  cvt(x, xb, (size_t)BS * 1024);

  dim3 blk(256);
  // QKV projection: qkv[8192,3072] = xb @ wqkv^T
  gemm_bf16_nt<false, false, false, false, true><<<dim3(3072 / BN, BS / BM), blk, 0, stream>>>(
      xb, 1024, wqkv, nullptr, nullptr, nullptr, qkv, 3072);
  // attention
  flash_attn<<<dim3(SEQ / 64, BATCH * NUM_HEADS), dim3(128), 0, stream>>>(qkv, zb);
  // attn out projection + residual: x1 = x + z @ W_O^T
  gemm_bf16_nt<false, false, true, true, true><<<dim3(1024 / BN, BS / BM), blk, 0, stream>>>(
      zb, 1024, wo, nullptr, x, x1f, x1b, 1024);
  // MLP in: h = gelu(x1 @ W_in^T + b_in)
  gemm_bf16_nt<true, true, false, false, true><<<dim3(4096 / BN, BS / BM), blk, 0, stream>>>(
      x1b, 1024, win, b_in, nullptr, nullptr, hb, 4096);
  // MLP out: out = x1 + h @ W_out^T + b_out
  gemm_bf16_nt<true, false, true, true, false><<<dim3(1024 / BN, BS / BM), blk, 0, stream>>>(
      hb, 4096, wout, b_out, x1f, out, nullptr, 1024);
}
